// MultiHeadGroupAttn_69836168233089
// MI455X (gfx1250) — matrix operand order repaired and checked by reading, not compiled, not run on hardware
//
#include <hip/hip_runtime.h>
#include <hip/hip_bf16.h>

// MultiHeadGroupAttn for MI455X (gfx1250), wave32 + WMMA f16.
//
// Single-query-per-head algebra: K/V projections are folded through the
// attention (qk = Q_h@Wk_h, ctx_h = (attn@G)@Wv_h^T), so `group` (512MB)
// is streamed exactly once -> memory bound (~22us roofline at 23.3 TB/s).
// All matmuls use v_wmma_f32_16x16x32_f16; qk staging uses the gfx1250
// async global->LDS DMA path (ASYNCcnt) when the builtin is available.

typedef __attribute__((ext_vector_type(16))) _Float16 v16h;
typedef __attribute__((ext_vector_type(8)))  _Float16 v8h;
typedef __attribute__((ext_vector_type(4)))  _Float16 v4h;
typedef __attribute__((ext_vector_type(8)))  float    v8f;
typedef __attribute__((ext_vector_type(4)))  float    v4f;
typedef int v4i __attribute__((vector_size(16)));  // matches builtin param type

#if defined(__has_builtin)
#if __has_builtin(__builtin_amdgcn_global_load_async_to_lds_b128)
#define ASYNC_LDS 1
#endif
#endif

#define DEVINL __device__ __forceinline__

static constexpr int B  = 2048;
static constexpr int N  = 64;
static constexpr int H  = 1024;
static constexpr int NH = 16;
// HD = 64, 1/sqrt(HD) = 1/8

DEVINL void wait_async_zero() {
#if defined(__has_builtin)
#if __has_builtin(__builtin_amdgcn_s_wait_asynccnt)
  __builtin_amdgcn_s_wait_asynccnt(0);
#else
  asm volatile("s_wait_asynccnt 0" ::: "memory");
#endif
#else
  asm volatile("s_wait_asynccnt 0" ::: "memory");
#endif
}

// ---- WMMA fragment helpers (layouts per CDNA5 ISA 7.12.2) -------------------
// A (16x32 f16): lane m (0-15) k=0..7,16..23 ; lane m+16 k=8..15,24..31
DEVINL v16h frag_a(const _Float16* base, int stride, int row0, int k0, int lane) {
  const int r  = row0 + (lane & 15);
  const int hi = (lane >> 4) * 8;
  const _Float16* p = base + r * stride + k0 + hi;
  v8h lo = *(const v8h*)(p);
  v8h hh = *(const v8h*)(p + 16);
  return __builtin_shufflevector(lo, hh, 0,1,2,3,4,5,6,7,8,9,10,11,12,13,14,15);
}
// B (32x16 f16): lane n (0-15) k=0..7,16..23 ; lane n+16 k=8..15,24..31
DEVINL v16h frag_b(const _Float16* base, int stride, int row0, int k0, int lane) {
  const int r  = row0 + (lane & 15);
  const int ks = k0 + (lane >> 4) * 8;
  const _Float16* p = base + r * stride + ks;
  v8h lo = *(const v8h*)(p);
  v8h hh = *(const v8h*)(p + 16);
  return __builtin_shufflevector(lo, hh, 0,1,2,3,4,5,6,7,8,9,10,11,12,13,14,15);
}
DEVINL v8f wmma_f16(v16h a, v16h b, v8f c) {
  return __builtin_amdgcn_wmma_f32_16x16x32_f16(false, a, false, b, (short)0, c,
                                                false, false);
}

// ---- f32 -> f16 conversion --------------------------------------------------
__global__ __launch_bounds__(256) void cvt_f32_to_f16(
    const float* __restrict__ src, _Float16* __restrict__ dst, int n4) {
  int i = blockIdx.x * 256 + threadIdx.x;
  if (i < n4) {
    v4f v = ((const v4f*)src)[i];
    v4h h = { (_Float16)v[0], (_Float16)v[1], (_Float16)v[2], (_Float16)v[3] };
    ((v4h*)dst)[i] = h;
  }
}

// 1024x1024 transpose + convert: dst[i*1024+j] = (f16)src[j*1024+i]
__global__ __launch_bounds__(256) void cvt_transpose_1024(
    const float* __restrict__ src, _Float16* __restrict__ dst) {
  __shared__ _Float16 t[64 * 72];
  const int tid = threadIdx.x;
  const int bx = blockIdx.x & 15;   // src column block
  const int by = blockIdx.x >> 4;   // src row block
  const int r  = tid >> 2;
  const int cs = (tid & 3) * 16;
  const v4f* sp = (const v4f*)(src + (size_t)(by * 64 + r) * 1024 + bx * 64 + cs);
#pragma unroll
  for (int j = 0; j < 4; ++j) {
    v4f v = sp[j];
    t[r * 72 + cs + j * 4 + 0] = (_Float16)v[0];
    t[r * 72 + cs + j * 4 + 1] = (_Float16)v[1];
    t[r * 72 + cs + j * 4 + 2] = (_Float16)v[2];
    t[r * 72 + cs + j * 4 + 3] = (_Float16)v[3];
  }
  __syncthreads();
  _Float16* dp = dst + (size_t)(bx * 64 + r) * 1024 + by * 64 + cs;
#pragma unroll
  for (int j = 0; j < 16; ++j) dp[j] = t[(cs + j) * 72 + r];
}

// ---- Generic WMMA GEMM: C[m, n] = sum_k A[m, kOffA+k] * Bw[nOffB+n, kOffB+k]
// Tile: 128(M) x 64(N) per block, 8 waves of 32x32, BK = 32.
template<bool F32OUT>
__global__ __launch_bounds__(256) void gemm_tn(
    const _Float16* __restrict__ A, int lda, int aKoffZ,
    const _Float16* __restrict__ Bw, int ldb, int bKoffZ, int bNoffZ,
    void* __restrict__ Cv, int ldc, int cNoffZ, int K) {
  __shared__ _Float16 sA[128 * 40];
  __shared__ _Float16 sB[64 * 40];
  const int tid  = threadIdx.x;
  const int lane = tid & 31;
  const int w    = tid >> 5;
  const int z    = blockIdx.z;
  const long aRow0 = (long)blockIdx.x * 128;
  const long bRow0 = (long)blockIdx.y * 64 + (long)bNoffZ * z;
  const int  aKoff = aKoffZ * z;
  const int  bKoff = bKoffZ * z;
  const int  wm = (w & 3) * 32;
  const int  wn = (w >> 2) * 32;
  v8f acc[2][2] = {};

  const int ar = tid >> 1, as = (tid & 1) * 16;
  for (int kt = 0; kt < K; kt += 32) {
    const uint4* ap = (const uint4*)(A + (aRow0 + ar) * (long)lda + aKoff + kt + as);
    uint4 a0 = ap[0], a1 = ap[1];
    uint4 b0 = {}, b1 = {};
    if (tid < 128) {
      const uint4* bp =
          (const uint4*)(Bw + (bRow0 + (tid >> 1)) * (long)ldb + bKoff + kt + as);
      b0 = bp[0]; b1 = bp[1];
    }
    __syncthreads();
    *(uint4*)(sA + ar * 40 + as)     = a0;
    *(uint4*)(sA + ar * 40 + as + 8) = a1;
    if (tid < 128) {
      *(uint4*)(sB + (tid >> 1) * 40 + as)     = b0;
      *(uint4*)(sB + (tid >> 1) * 40 + as + 8) = b1;
    }
    __syncthreads();
    v16h bf0 = frag_b(sB, 40, wn,      0, lane);
    v16h bf1 = frag_b(sB, 40, wn + 16, 0, lane);
#pragma unroll
    for (int im = 0; im < 2; ++im) {
      v16h af = frag_a(sA, 40, wm + im * 16, 0, lane);
      acc[im][0] = wmma_f16(af, bf0, acc[im][0]);
      acc[im][1] = wmma_f16(af, bf1, acc[im][1]);
    }
  }
  const int hi = lane >> 4, nl = lane & 15;
#pragma unroll
  for (int im = 0; im < 2; ++im)
#pragma unroll
    for (int in = 0; in < 2; ++in) {
      const long n = (long)blockIdx.y * 64 + (long)cNoffZ * z + wn + in * 16 + nl;
#pragma unroll
      for (int r = 0; r < 8; ++r) {
        const long m = aRow0 + wm + im * 16 + r + 8 * hi;
        if (F32OUT) ((float*)Cv)[m * (long)ldc + n] = acc[im][in][r];
        else        ((_Float16*)Cv)[m * (long)ldc + n] = (_Float16)acc[im][in][r];
      }
    }
}

// ---- Fused attention: one workgroup per batch row b -------------------------
// LDS (halves): sG[64][1032] rowmajor G, sGT[1024][72] transposed G,
//               sQK[16][1032], sAttn16[16][72], sLogits 64x16 f32. ~312KB.
static constexpr int GS  = 1032;  // sG / sQK row stride (halves)
static constexpr int TS  = 72;    // sGT / sAttn row stride (halves)
static constexpr size_t ATTN_LDS =
    (size_t)(64 * GS + 1024 * TS + 16 * GS + 16 * TS) * 2 + 64 * 16 * 4;

__global__ __launch_bounds__(256) void attn_kernel(
    const float* __restrict__ group, const unsigned char* __restrict__ mask,
    const float* __restrict__ gtemp, const _Float16* __restrict__ qk16,
    _Float16* __restrict__ gbar16, float* __restrict__ attn_out) {
  extern __shared__ _Float16 smem[];
  _Float16* sG  = smem;                 // [64][GS]
  _Float16* sGT = sG  + 64 * GS;        // [1024][TS]
  _Float16* sQK = sGT + 1024 * TS;      // [16][GS]
  _Float16* sAt = sQK + 16 * GS;        // [16][TS]
  float*    sLg = (float*)(sAt + 16 * TS);  // [64][16]

  const int tid  = threadIdx.x;
  const int lane = tid & 31;
  const int w    = tid >> 5;
  const int b    = blockIdx.x;

  // stage qk[b] (16x1024 f16) into LDS -- async DMA path when available
  {
    const _Float16* qg = qk16 + (size_t)b * 16384;
    for (int it = tid; it < 2048; it += 256) {
      const int hoff = it * 8;
#ifdef ASYNC_LDS
      __builtin_amdgcn_global_load_async_to_lds_b128(
          (v4i*)(qg + hoff),
          (v4i*)(sQK + (hoff >> 10) * GS + (hoff & 1023)), 0, 0);
#else
      uint4 v = *(const uint4*)(qg + hoff);
      *(uint4*)(sQK + (hoff >> 10) * GS + (hoff & 1023)) = v;
#endif
    }
  }
  // stream group[b] (64x1024 f32) once, non-temporal (single use; keep L2 for
  // qk/gbar).  4 rows per step; 4x4 register transpose -> v4h stores to sGT.
  {
    const v4f* gp = (const v4f*)(group + (size_t)b * (N * H));
    const int i0 = tid * 4;
    for (int n0 = 0; n0 < N; n0 += 4) {
      v4f v0 = __builtin_nontemporal_load(gp + (n0 + 0) * 256 + tid);
      v4f v1 = __builtin_nontemporal_load(gp + (n0 + 1) * 256 + tid);
      v4f v2 = __builtin_nontemporal_load(gp + (n0 + 2) * 256 + tid);
      v4f v3 = __builtin_nontemporal_load(gp + (n0 + 3) * 256 + tid);
      v4h r0 = { (_Float16)v0[0], (_Float16)v0[1], (_Float16)v0[2], (_Float16)v0[3] };
      v4h r1 = { (_Float16)v1[0], (_Float16)v1[1], (_Float16)v1[2], (_Float16)v1[3] };
      v4h r2 = { (_Float16)v2[0], (_Float16)v2[1], (_Float16)v2[2], (_Float16)v2[3] };
      v4h r3 = { (_Float16)v3[0], (_Float16)v3[1], (_Float16)v3[2], (_Float16)v3[3] };
      *(v4h*)(sG + (n0 + 0) * GS + i0) = r0;
      *(v4h*)(sG + (n0 + 1) * GS + i0) = r1;
      *(v4h*)(sG + (n0 + 2) * GS + i0) = r2;
      *(v4h*)(sG + (n0 + 3) * GS + i0) = r3;
      v4h c0 = { r0[0], r1[0], r2[0], r3[0] };
      v4h c1 = { r0[1], r1[1], r2[1], r3[1] };
      v4h c2 = { r0[2], r1[2], r2[2], r3[2] };
      v4h c3 = { r0[3], r1[3], r2[3], r3[3] };
      *(v4h*)(sGT + (i0 + 0) * TS + n0) = c0;
      *(v4h*)(sGT + (i0 + 1) * TS + n0) = c1;
      *(v4h*)(sGT + (i0 + 2) * TS + n0) = c2;
      *(v4h*)(sGT + (i0 + 3) * TS + n0) = c3;
    }
  }
#ifdef ASYNC_LDS
  wait_async_zero();
#endif
  __syncthreads();

  // phase 1: logits^T[n][h] = sum_i G[n,i]*qk[h,i]  (WMMA M=64,N=16,K=1024)
  if (w < 4) {
    const int ntile = w * 16;
    v8f c = {};
#pragma unroll 4
    for (int kt = 0; kt < H; kt += 32) {
      v16h af = frag_a(sG, GS, ntile, kt, lane);
      v16h bf = frag_b(sQK, GS, 0, kt, lane);
      c = wmma_f16(af, bf, c);
    }
    const int h = lane & 15, hi = lane >> 4;
#pragma unroll
    for (int r = 0; r < 8; ++r) sLg[(ntile + r + 8 * hi) * 16 + h] = c[r];
  }
  __syncthreads();

  // softmax per head (mask, 1/sqrt(64), 1/temp)
  if (tid < NH) {
    const int h = tid;
    const float inv = __builtin_amdgcn_rcpf(8.0f * gtemp[b]);
    float mx = -3.0e38f;
    for (int n = 0; n < N; ++n) {
      float l = sLg[n * 16 + h] * inv;
      l = mask[(size_t)b * N + n] ? l : -1.0e9f;
      sLg[n * 16 + h] = l;
      mx = fmaxf(mx, l);
    }
    float s = 0.0f;
    for (int n = 0; n < N; ++n) {
      float e = __expf(sLg[n * 16 + h] - mx);
      sLg[n * 16 + h] = e;
      s += e;
    }
    const float rs = __builtin_amdgcn_rcpf(s);
    float* ao = attn_out + ((size_t)b * NH + h) * N;
    for (int n = 0; n < N; ++n) {
      float a = sLg[n * 16 + h] * rs;
      ao[n] = a;
      sAt[h * TS + n] = (_Float16)a;
    }
  }
  __syncthreads();

  // phase 2: gbar[h][i] = sum_n attn[h][n]*G[n][i]  (WMMA M=16,N=1024,K=64)
  {
    v16h a0 = frag_a(sAt, TS, 0, 0,  lane);
    v16h a1 = frag_a(sAt, TS, 0, 32, lane);
    const int nl = lane & 15, hi = lane >> 4;
    for (int it = w * 16; it < H; it += 128) {
      v8f c = {};
      c = wmma_f16(a0, frag_b(sGT, TS, it, 0,  lane), c);
      c = wmma_f16(a1, frag_b(sGT, TS, it, 32, lane), c);
      const int i = it + nl;
#pragma unroll
      for (int r = 0; r < 8; ++r)
        gbar16[((size_t)b * NH + (r + 8 * hi)) * H + i] = (_Float16)c[r];
    }
  }
}

extern "C" void kernel_launch(void* const* d_in, const int* in_sizes, int n_in,
                              void* d_out, int out_size, void* d_ws, size_t ws_size,
                              hipStream_t stream) {
  const float* q_src = (const float*)d_in[0];
  const float* group = (const float*)d_in[1];
  const unsigned char* mask = (const unsigned char*)d_in[2];  // jax bool = 1B
  const float* gtemp = (const float*)d_in[3];
  const float* Wq = (const float*)d_in[4];
  const float* Wk = (const float*)d_in[5];
  const float* Wv = (const float*)d_in[6];
  const float* Wo = (const float*)d_in[7];

  float* out      = (float*)d_out;                     // [2048,1024]
  float* attn_out = out + (size_t)B * H;               // [2048,16,64]

  // workspace carve-up (halves), ~148MB total
  _Float16* ws     = (_Float16*)d_ws;
  _Float16* q16    = ws;                    // 2M
  _Float16* Wq16   = q16    + 2097152;      // 1M
  _Float16* WkT16  = Wq16   + 1048576;      // 1M (Wk transposed)
  _Float16* Wv16   = WkT16  + 1048576;      // 1M
  _Float16* Wo16   = Wv16   + 1048576;      // 1M
  _Float16* Q16    = Wo16   + 1048576;      // 2M
  _Float16* qk16   = Q16    + 2097152;      // 32M: [B][NH][H]
  _Float16* gbar16 = qk16   + 33554432;     // 32M: [B][NH][H]
  _Float16* ctx16  = gbar16 + 33554432;     // 2M:  [B][H]

  cvt_f32_to_f16<<<2048, 256, 0, stream>>>(q_src, q16, 524288);
  cvt_f32_to_f16<<<1024, 256, 0, stream>>>(Wq, Wq16, 262144);
  cvt_transpose_1024<<<256, 256, 0, stream>>>(Wk, WkT16);
  cvt_f32_to_f16<<<1024, 256, 0, stream>>>(Wv, Wv16, 262144);
  cvt_f32_to_f16<<<1024, 256, 0, stream>>>(Wo, Wo16, 262144);

  // Q = q_src @ Wq^T                         (M=2048, N=1024, K=1024)
  gemm_tn<false><<<dim3(16, 16, 1), 256, 0, stream>>>(
      q16, 1024, 0, Wq16, 1024, 0, 0, Q16, 1024, 0, 1024);
  // qk[:,h,:] = Q[:,64h:64h+64] @ WkT[:,64h:64h+64]^T   (K=64, z=head)
  gemm_tn<false><<<dim3(16, 16, 16), 256, 0, stream>>>(
      Q16, 1024, 64, WkT16, 1024, 64, 0, qk16, 16384, 1024, 64);

  (void)hipFuncSetAttribute((const void*)attn_kernel,
                            hipFuncAttributeMaxDynamicSharedMemorySize,
                            (int)ATTN_LDS);
  attn_kernel<<<B, 256, ATTN_LDS, stream>>>(group, mask, gtemp, qk16, gbar16,
                                            attn_out);

  // ctx[:,64h:64h+64] = gbar[:,h,:] @ Wv[64h:64h+64,:]^T  (N=64, K=1024)
  gemm_tn<false><<<dim3(16, 1, 16), 256, 0, stream>>>(
      gbar16, 16384, 1024, Wv16, 1024, 0, 64, ctx16, 1024, 64, 1024);
  // out = ctx @ Wo^T (f32 out)
  gemm_tn<true><<<dim3(16, 16, 1), 256, 0, stream>>>(
      ctx16, 1024, 0, Wo16, 1024, 0, 0, d_out, 1024, 0, 1024);
}
